// RevisedSpatialAttention_17841294147823
// MI455X (gfx1250) — matrix operand order repaired and checked by reading, not compiled, not run on hardware
//
#include <hip/hip_runtime.h>

// ---------------------------------------------------------------------------
// DANet dual attention (CAM + flash-PAM) for MI455X / gfx1250.
// All GEMMs via v_wmma_f32_16x16x32_f16 (f16 in, f32 accumulate), wave32.
// PAM: 8-wave workgroups (128 rows), K/V j-tiles staged cooperatively into
// LDS via global_load_async_to_lds_b128 (ASYNCcnt) and shared by all 8
// waves -> 8x less L2 fragment traffic. The 4096x4096 attention matrix is
// never materialized. Workspace requirement: ~92 MB.
// ---------------------------------------------------------------------------

typedef _Float16 v8h  __attribute__((ext_vector_type(8)));
typedef _Float16 v16h __attribute__((ext_vector_type(16)));
typedef float    v8f  __attribute__((ext_vector_type(8)));
typedef int      v4i  __attribute__((ext_vector_type(4)));

typedef v4i __attribute__((address_space(1)))* gas_v4i_ptr;
typedef v4i __attribute__((address_space(3)))* las_v4i_ptr;

#define DEVFN static __device__ __forceinline__

constexpr int B  = 8;
constexpr int C  = 256;
constexpr int H  = 64;
constexpr int W  = 64;
constexpr int N  = H * W;   // 4096
constexpr int C8 = C / 8;   // 32

union Frag16 { v16h v; struct { v8h lo, hi; } h; };

DEVFN v8f wmma_f16(v16h a, v16h b, v8f c) {
  // (neg_a, A, neg_b, B, c_mod, C, reuse_a, reuse_b)
  return __builtin_amdgcn_wmma_f32_16x16x32_f16(false, a, false, b, (short)0, c,
                                                false, false);
}

// A-fragment (M=16, K=32) from row-major [rows x ld] memory.
// ISA layout: lanes 0-15 hold K {0..7,16..23}, lanes 16-31 hold K {8..15,24..31}.
DEVFN v16h load_a(const _Float16* base, int ld, int row0, int k0) {
  int lane = threadIdx.x & 31;
  const _Float16* p =
      base + (size_t)(row0 + (lane & 15)) * ld + k0 + ((lane >> 4) << 3);
  Frag16 f;
  f.h.lo = *(const v8h*)p;
  f.h.hi = *(const v8h*)(p + 16);
  return f.v;
}

// B-fragment (K=32, N=16) from row-major [cols x ld] memory (i.e. B^T rows).
// ISA layout: lanes 0-15 hold K {0..7,16..23}, lanes 16-31 hold K {8..15,24..31}.
DEVFN v16h load_b(const _Float16* base, int ld, int col0, int k0) {
  int lane = threadIdx.x & 31;
  const _Float16* p =
      base + (size_t)(col0 + (lane & 15)) * ld + k0 + ((lane >> 4) << 3);
  Frag16 f;
  f.h.lo = *(const v8h*)p;
  f.h.hi = *(const v8h*)(p + 16);
  return f.v;
}

// ---------------------------------------------------------------------------
// Async global->LDS copy (16B granules). Uses the gfx1250 ASYNCcnt path when
// the toolchain exposes it; otherwise a plain load + ds_store fallback.
// ---------------------------------------------------------------------------
#if __has_builtin(__builtin_amdgcn_global_load_async_to_lds_b128)
#define ASYNC_LDS 1
#else
#define ASYNC_LDS 0
#endif

DEVFN void copy16_g2l(const _Float16* __restrict__ g, _Float16* l) {
#if ASYNC_LDS
  // Builtin signature: (int4 AS1* src, int4 AS3* dst, imm offset, imm cpol).
  // Flat addresses of __shared__ objects carry the LDS byte offset in [31:0].
  __builtin_amdgcn_global_load_async_to_lds_b128(
      (gas_v4i_ptr)(unsigned long long)g,
      (las_v4i_ptr)(unsigned int)(unsigned long long)l, 0, 0);
#else
  *(v8h*)l = *(const v8h*)g;
#endif
}

DEVFN void wait_async_copies() {
#if ASYNC_LDS
#if __has_builtin(__builtin_amdgcn_s_wait_asynccnt)
  __builtin_amdgcn_s_wait_asynccnt(0);
#else
  asm volatile("s_wait_asynccnt 0x0" ::: "memory");
#endif
#endif
}

// ---------------------------------------------------------------------------
// 1) Precision conversion: x -> Xh [B,C,N] f16 and XhT [B,N,C] f16
// ---------------------------------------------------------------------------
__global__ void k_convert_x(const float* __restrict__ x,
                            _Float16* __restrict__ Xh,
                            _Float16* __restrict__ XhT) {
  const int total = B * C * N;
  for (int i = blockIdx.x * blockDim.x + threadIdx.x; i < total;
       i += gridDim.x * blockDim.x) {
    float v = x[i];
    int n = i % N;
    int c = (i / N) % C;
    int b = i / (N * C);
    Xh[i] = (_Float16)v;
    XhT[((size_t)b * N + n) * C + c] = (_Float16)v;
  }
}

__global__ void k_convert_w(const float* __restrict__ wq,
                            const float* __restrict__ wk,
                            const float* __restrict__ wv,
                            _Float16* __restrict__ wqh,
                            _Float16* __restrict__ wkh,
                            _Float16* __restrict__ wvh) {
  int i = blockIdx.x * blockDim.x + threadIdx.x;
  if (i < C8 * C) {
    wqh[i] = (_Float16)wq[i];
    wkh[i] = (_Float16)wk[i];
  }
  if (i < C * C) wvh[i] = (_Float16)wv[i];
}

// ---------------------------------------------------------------------------
// 2) CAM energy: energy[b] = X X^T  (M=C, Nout=C, K=N)
// ---------------------------------------------------------------------------
__global__ void k_cam_energy(const _Float16* __restrict__ Xh,
                             float* __restrict__ energy) {
  int b  = blockIdx.y;
  int tm = (blockIdx.x / (C / 16)) * 16;
  int tn = (blockIdx.x % (C / 16)) * 16;
  const _Float16* Xb = Xh + (size_t)b * C * N;
  v8f acc = {};
#pragma unroll 4
  for (int k0 = 0; k0 < N; k0 += 32)
    acc = wmma_f16(load_a(Xb, N, tm, k0), load_b(Xb, N, tn, k0), acc);
  int lane  = threadIdx.x & 31;
  int nn    = tn + (lane & 15);
  int mbase = tm + ((lane >> 4) << 3);
  float* e  = energy + (size_t)b * C * C;
#pragma unroll
  for (int v = 0; v < 8; ++v) e[(size_t)(mbase + v) * C + nn] = acc[v];
}

// ---------------------------------------------------------------------------
// 3) Row softmax over C=256, one wave per row, f16 output
// ---------------------------------------------------------------------------
__global__ void k_softmax_c(const float* __restrict__ energy,
                            _Float16* __restrict__ attn) {
  int row = blockIdx.x;  // B*C rows
  const float* e = energy + (size_t)row * C;
  int lane = threadIdx.x & 31;
  float vals[8];
  float mx = -3.0e38f;
#pragma unroll
  for (int i = 0; i < 8; ++i) {
    vals[i] = e[lane + 32 * i];
    mx = fmaxf(mx, vals[i]);
  }
#pragma unroll
  for (int m = 16; m >= 1; m >>= 1) mx = fmaxf(mx, __shfl_xor(mx, m, 32));
  float s = 0.f;
#pragma unroll
  for (int i = 0; i < 8; ++i) {
    vals[i] = __expf(vals[i] - mx);
    s += vals[i];
  }
#pragma unroll
  for (int m = 16; m >= 1; m >>= 1) s += __shfl_xor(s, m, 32);
  float inv = 1.0f / s;
  _Float16* a = attn + (size_t)row * C;
#pragma unroll
  for (int i = 0; i < 8; ++i) a[lane + 32 * i] = (_Float16)(vals[i] * inv);
}

// ---------------------------------------------------------------------------
// 4) camBuf = gamma * (attn @ X) + x   (M=C, Nout=N, K=C)
// ---------------------------------------------------------------------------
__global__ void k_cam_apply(const _Float16* __restrict__ attn,
                            const _Float16* __restrict__ XhT,
                            const float* __restrict__ x,
                            const float* __restrict__ gamma,
                            float* __restrict__ camBuf) {
  int b  = blockIdx.y;
  int tm = (blockIdx.x / (N / 16)) * 16;  // channel tile
  int tn = (blockIdx.x % (N / 16)) * 16;  // spatial tile
  const _Float16* A  = attn + (size_t)b * C * C;
  const _Float16* Bt = XhT + (size_t)b * N * C;
  v8f acc = {};
#pragma unroll
  for (int k0 = 0; k0 < C; k0 += 32)
    acc = wmma_f16(load_a(A, C, tm, k0), load_b(Bt, C, tn, k0), acc);
  float g   = gamma[0];
  int lane  = threadIdx.x & 31;
  int nn    = tn + (lane & 15);
  int mbase = tm + ((lane >> 4) << 3);
  const float* xb = x + (size_t)b * C * N;
  float* cb = camBuf + (size_t)b * C * N;
#pragma unroll
  for (int v = 0; v < 8; ++v) {
    size_t idx = (size_t)(mbase + v) * N + nn;
    cb[idx] = g * acc[v] + xb[idx];
  }
}

// ---------------------------------------------------------------------------
// 5a) Qt/Kt [N x 32] = X^T @ Wq^T + bq  (M=N, Nout=C8, K=C)
// ---------------------------------------------------------------------------
__global__ void k_proj_qk(const _Float16* __restrict__ XhT,
                          const _Float16* __restrict__ wqh,
                          const _Float16* __restrict__ wkh,
                          const float* __restrict__ bq,
                          const float* __restrict__ bk,
                          _Float16* __restrict__ Qt,
                          _Float16* __restrict__ Kt) {
  int b     = blockIdx.y;
  bool isK  = blockIdx.z != 0;
  const _Float16* Wm = isK ? wkh : wqh;
  const float* bias  = isK ? bk : bq;
  _Float16* Out = (isK ? Kt : Qt) + (size_t)b * N * C8;
  int tm = (blockIdx.x / (C8 / 16)) * 16;  // spatial rows
  int tn = (blockIdx.x % (C8 / 16)) * 16;  // out channels
  const _Float16* A = XhT + (size_t)b * N * C;
  v8f acc = {};
#pragma unroll
  for (int k0 = 0; k0 < C; k0 += 32)
    acc = wmma_f16(load_a(A, C, tm, k0), load_b(Wm, C, tn, k0), acc);
  int lane  = threadIdx.x & 31;
  int nn    = tn + (lane & 15);
  int mbase = tm + ((lane >> 4) << 3);
  float bb = bias[nn];
#pragma unroll
  for (int v = 0; v < 8; ++v)
    Out[(size_t)(mbase + v) * C8 + nn] = (_Float16)(acc[v] + bb);
}

// ---------------------------------------------------------------------------
// 5b) Vh [C x N] = Wv @ X + bv   (M=C, Nout=N, K=C)
// ---------------------------------------------------------------------------
__global__ void k_proj_v(const _Float16* __restrict__ wvh,
                         const _Float16* __restrict__ XhT,
                         const float* __restrict__ bv,
                         _Float16* __restrict__ Vh) {
  int b  = blockIdx.y;
  int tm = (blockIdx.x / (N / 16)) * 16;  // channel rows
  int tn = (blockIdx.x % (N / 16)) * 16;  // spatial cols
  const _Float16* Bt = XhT + (size_t)b * N * C;
  v8f acc = {};
#pragma unroll
  for (int k0 = 0; k0 < C; k0 += 32)
    acc = wmma_f16(load_a(wvh, C, tm, k0), load_b(Bt, C, tn, k0), acc);
  int lane  = threadIdx.x & 31;
  int nn    = tn + (lane & 15);
  int mbase = tm + ((lane >> 4) << 3);
  _Float16* Vb = Vh + (size_t)b * C * N;
#pragma unroll
  for (int v = 0; v < 8; ++v)
    Vb[(size_t)(mbase + v) * N + nn] = (_Float16)(acc[v] + bv[mbase + v]);
}

// ---------------------------------------------------------------------------
// 6) Flash PAM, 8 waves per workgroup (128 rows of i per WG).
//    Per j-chunk of 32 columns: K tile (32x32) and V tile (256x32) are staged
//    into LDS once (async-to-LDS), then all 8 waves consume them.
//    out[b,c,i] = camBuf[b,c,i] + sum_j softmax_j(q_i . k_j) * V[c,j]
// ---------------------------------------------------------------------------
__global__ void k_flash_pam(const _Float16* __restrict__ Qt,
                            const _Float16* __restrict__ Kt,
                            const _Float16* __restrict__ Vh,
                            const float* __restrict__ camBuf,
                            float* __restrict__ out) {
  __shared__ _Float16 Ktile[32 * 32];        // [j in chunk][d], ld = 32
  __shared__ _Float16 Vtile[C * 32];         // [c][j in chunk], ld = 32
  __shared__ _Float16 Ptile[8][16 * 32];     // per-wave P tile, ld = 32

  int b    = blockIdx.y;
  int tid  = threadIdx.x;
  int lane = tid & 31;
  int wv   = tid >> 5;                       // wave id 0..7
  int i0   = blockIdx.x * 128 + wv * 16;     // this wave's 16 query rows
  int half = (lane >> 4) << 3;               // row offset of D-layout rows

  const _Float16* Qb = Qt + (size_t)b * N * C8;
  const _Float16* Kb = Kt + (size_t)b * N * C8;
  const _Float16* Vb = Vh + (size_t)b * C * N;
  _Float16* Pw = &Ptile[wv][0];

  // Q fragment: head dim 32 == exactly one WMMA K slice
  v16h qa = load_a(Qb, C8, i0, 0);

  v8f zero = {};
  v8f O[16];  // 16 rows x 256 channels of f32 accumulators (128 VGPRs)
#pragma unroll
  for (int c = 0; c < 16; ++c) O[c] = zero;

  float m_run[8], l_run[8];
#pragma unroll
  for (int v = 0; v < 8; ++v) {
    m_run[v] = -3.0e38f;
    l_run[v] = 0.0f;
  }

  // 16B granules to stage per j-chunk: V = 1024, K = 128
  constexpr int VCH = (C * 32) / 8;          // 1024
  constexpr int TOT = VCH + (32 * 32) / 8;   // 1152

  for (int j0 = 0; j0 < N; j0 += 32) {
    // ---- cooperative staging of K/V tiles into LDS --------------------
    for (int ch = tid; ch < TOT; ch += 256) {
      if (ch < VCH) {
        int c = ch >> 2, seg = (ch & 3) * 8;
        copy16_g2l(Vb + (size_t)c * N + j0 + seg, &Vtile[c * 32 + seg]);
      } else {
        int e = ch - VCH;
        int jj = e >> 2, seg = (e & 3) * 8;
        copy16_g2l(Kb + (size_t)(j0 + jj) * C8 + seg, &Ktile[jj * 32 + seg]);
      }
    }
    wait_async_copies();
    __syncthreads();

    // ---- S = Q K^T for this wave's 16 rows ----------------------------
    v16h kb0 = load_b(Ktile, 32, 0, 0);
    v16h kb1 = load_b(Ktile, 32, 16, 0);
    v8f s0 = wmma_f16(qa, kb0, zero);
    v8f s1 = wmma_f16(qa, kb1, zero);

    // per-row block max (rows live at VGPR index within each 16-lane half)
    float bmax[8];
#pragma unroll
    for (int v = 0; v < 8; ++v) bmax[v] = fmaxf(s0[v], s1[v]);
#pragma unroll
    for (int m = 8; m >= 1; m >>= 1)
#pragma unroll
      for (int v = 0; v < 8; ++v)
        bmax[v] = fmaxf(bmax[v], __shfl_xor(bmax[v], m, 32));

    float scale[8], psum[8];
#pragma unroll
    for (int v = 0; v < 8; ++v) {
      float mn = fmaxf(m_run[v], bmax[v]);
      scale[v] = __expf(m_run[v] - mn);
      m_run[v] = mn;
      float p0 = __expf(s0[v] - mn);
      float p1 = __expf(s1[v] - mn);
      int mrow = v + half;
      Pw[mrow * 32 + (lane & 15)]      = (_Float16)p0;
      Pw[mrow * 32 + 16 + (lane & 15)] = (_Float16)p1;
      psum[v] = p0 + p1;
    }
#pragma unroll
    for (int m = 8; m >= 1; m >>= 1)
#pragma unroll
      for (int v = 0; v < 8; ++v) psum[v] += __shfl_xor(psum[v], m, 32);
#pragma unroll
    for (int v = 0; v < 8; ++v) l_run[v] = l_run[v] * scale[v] + psum[v];

    // rescale accumulators by per-row scale
#pragma unroll
    for (int c = 0; c < 16; ++c)
#pragma unroll
      for (int v = 0; v < 8; ++v) O[c][v] *= scale[v];

    // ---- O += P @ V^T: V fragments come from the shared tile ----------
    v16h pa = load_a(Pw, 32, 0, 0);  // re-layout P as WMMA A operand
#pragma unroll 4
    for (int c = 0; c < 16; ++c) {
      v16h vb = load_b(Vtile, 32, c * 16, 0);
      O[c] = wmma_f16(pa, vb, O[c]);
    }
    __syncthreads();  // tiles fully consumed before next chunk's staging
  }

  // finalize: out = camBuf + O / l
  float inv[8];
#pragma unroll
  for (int v = 0; v < 8; ++v) inv[v] = 1.0f / l_run[v];
#pragma unroll
  for (int c = 0; c < 16; ++c) {
    int cc = c * 16 + (lane & 15);
    const float* cs = camBuf + ((size_t)b * C + cc) * N;
    float* os = out + ((size_t)b * C + cc) * N;
#pragma unroll
    for (int v = 0; v < 8; ++v) {
      int ii = i0 + half + v;
      os[ii] = cs[ii] + O[c][v] * inv[v];
    }
  }
}

// ---------------------------------------------------------------------------
extern "C" void kernel_launch(void* const* d_in, const int* in_sizes, int n_in,
                              void* d_out, int out_size, void* d_ws,
                              size_t ws_size, hipStream_t stream) {
  (void)in_sizes; (void)n_in; (void)out_size; (void)ws_size;
  const float* x     = (const float*)d_in[0];
  const float* gamma = (const float*)d_in[1];
  const float* wq    = (const float*)d_in[2];
  const float* bq    = (const float*)d_in[3];
  const float* wk    = (const float*)d_in[4];
  const float* bk    = (const float*)d_in[5];
  const float* wv    = (const float*)d_in[6];
  const float* bv    = (const float*)d_in[7];
  float* out = (float*)d_out;

  char* ws = (char*)d_ws;
  size_t off = 0;
  auto alloc = [&](size_t bytes) -> void* {
    off = (off + 255) & ~(size_t)255;
    void* p = ws + off;
    off += bytes;
    return p;
  };
  _Float16* Xh   = (_Float16*)alloc((size_t)B * C * N * 2);
  _Float16* XhT  = (_Float16*)alloc((size_t)B * N * C * 2);
  float*    ene  = (float*)   alloc((size_t)B * C * C * 4);
  _Float16* attn = (_Float16*)alloc((size_t)B * C * C * 2);
  _Float16* Qt   = (_Float16*)alloc((size_t)B * N * C8 * 2);
  _Float16* Kt   = (_Float16*)alloc((size_t)B * N * C8 * 2);
  _Float16* Vh   = (_Float16*)alloc((size_t)B * C * N * 2);
  float*    cam  = (float*)   alloc((size_t)B * C * N * 4);
  _Float16* wqh  = (_Float16*)alloc((size_t)C8 * C * 2);
  _Float16* wkh  = (_Float16*)alloc((size_t)C8 * C * 2);
  _Float16* wvh  = (_Float16*)alloc((size_t)C * C * 2);

  k_convert_x<<<4096, 256, 0, stream>>>(x, Xh, XhT);
  k_convert_w<<<(C * C + 255) / 256, 256, 0, stream>>>(wq, wk, wv, wqh, wkh, wvh);

  k_cam_energy<<<dim3((C / 16) * (C / 16), B), 32, 0, stream>>>(Xh, ene);
  k_softmax_c<<<B * C, 32, 0, stream>>>(ene, attn);
  k_cam_apply<<<dim3((C / 16) * (N / 16), B), 32, 0, stream>>>(attn, XhT, x, gamma, cam);

  k_proj_qk<<<dim3((N / 16) * (C8 / 16), B, 2), 32, 0, stream>>>(XhT, wqh, wkh, bq, bk, Qt, Kt);
  k_proj_v<<<dim3((C / 16) * (N / 16), B), 32, 0, stream>>>(wvh, XhT, bv, Vh);

  k_flash_pam<<<dim3(N / 128, B), 256, 0, stream>>>(Qt, Kt, Vh, cam, out);
}
